// STLinear_HopBiased_51393578664575
// MI455X (gfx1250) — matrix operand order repaired and checked by reading, not compiled, not run on hardware
//
#include <hip/hip_runtime.h>
#include <math.h>
#include <stdint.h>

// ---------------- problem constants ----------------
constexpr int B = 8, T = 12, NN = 400, IN_DIM = 3;
constexpr int OUT_STEPS = 3, OUT_DIM = 3;
constexpr int D_IN = 32, D_TOD = 32, D_DOW = 32, D_ADP = 80;
constexpr int DM = 176;           // D_MODEL
constexpr int FF = 256;
constexpr int H = 4, DH = 44, L = 3;
constexpr int BT = B * T;                 // 96
constexpr int M_ROWS = BT * NN;           // 38400  (multiple of 128)
constexpr int DMP = 192;                  // DM padded to mult of 64
constexpr int DHP = 64;                   // per-head K padded
constexpr int NP = 416;                   // nodes padded to mult of 32
constexpr int DHT = 48;                   // head-dim rows padded to mult of 16
constexpr int KOUT = T * DM;              // 2112
constexpr float SCALE = 0.15075567228888181f; // 1/sqrt(44)
constexpr float EPS = 1e-5f;

// ---------------- WMMA vector types ----------------
typedef __attribute__((ext_vector_type(16))) __bf16 v16bf;
typedef __attribute__((ext_vector_type(8)))  __bf16 v8bf;
typedef __attribute__((ext_vector_type(8)))  float  v8f;

static __device__ __forceinline__ v16bf cat8(v8bf a0, v8bf a1) {
  return __builtin_shufflevector(a0, a1, 0,1,2,3,4,5,6,7,8,9,10,11,12,13,14,15);
}

// ---------------- embedding ----------------
__global__ __launch_bounds__(256) void embed_kernel(
    const float* __restrict__ x, const int* __restrict__ tod,
    const int* __restrict__ dow, const float* __restrict__ in_W,
    const float* __restrict__ in_b, const float* __restrict__ tod_emb,
    const float* __restrict__ dow_emb, const float* __restrict__ adp,
    float* __restrict__ h)
{
  int i = blockIdx.x * 256 + threadIdx.x;
  if (i >= M_ROWS * DM) return;
  int d = i % DM;
  int row = i / DM;
  int n = row % NN;
  int bt = row / NN;
  int t = bt % T;
  float v;
  if (d < D_IN) {
    v = in_b[d];
    const float* xr = x + (size_t)row * IN_DIM;
    #pragma unroll
    for (int j = 0; j < IN_DIM; ++j) v += xr[j] * in_W[j * D_IN + d];
  } else if (d < D_IN + D_TOD) {
    v = tod_emb[tod[row] * D_TOD + (d - D_IN)];
  } else if (d < D_IN + D_TOD + D_DOW) {
    v = dow_emb[dow[row] * D_DOW + (d - D_IN - D_TOD)];
  } else {
    v = adp[((size_t)t * NN + n) * D_ADP + (d - D_IN - D_TOD - D_DOW)];
  }
  h[i] = v;
}

// ---------------- DLinear temporal block (in-place over h) ----------------
__global__ __launch_bounds__(256) void dlinear_kernel(
    float* __restrict__ h, const float* __restrict__ Ws, const float* __restrict__ bs,
    const float* __restrict__ Wt, const float* __restrict__ bt_)
{
  int i = blockIdx.x * 256 + threadIdx.x;
  if (i >= B * NN * DM) return;
  int d = i % DM;
  int n = (i / DM) % NN;
  int b = i / (DM * NN);
  size_t base = ((size_t)b * T * NN + n) * DM + d;
  const size_t stride = (size_t)NN * DM;
  float s[T], mean[T], res[T];
  #pragma unroll
  for (int t = 0; t < T; ++t) s[t] = h[base + t * stride];
  #pragma unroll
  for (int t = 0; t < T; ++t) {
    float acc = 0.f;
    #pragma unroll
    for (int j = -2; j <= 2; ++j) {
      int c = t + j; c = c < 0 ? 0 : (c > T - 1 ? T - 1 : c);
      acc += s[c];
    }
    mean[t] = acc * 0.2f;
    res[t] = s[t] - mean[t];
  }
  #pragma unroll
  for (int tt = 0; tt < T; ++tt) {
    float acc = bs[tt] + bt_[tt];
    #pragma unroll
    for (int t = 0; t < T; ++t)
      acc += res[t] * Ws[t * T + tt] + mean[t] * Wt[t * T + tt];
    h[base + tt * stride] = acc;
  }
}

// ---------------- f32 -> bf16 with K padding ----------------
__global__ __launch_bounds__(256) void convpad_kernel(
    const float* __restrict__ X, __bf16* __restrict__ Y, int Kin, int Kpad, int total)
{
  int i = blockIdx.x * 256 + threadIdx.x;
  if (i >= total) return;
  int k = i % Kpad;
  int m = i / Kpad;
  Y[i] = (__bf16)((k < Kin) ? X[(size_t)m * Kin + k] : 0.f);
}

// ---------------- weight transpose+convert: W[K][Nin] -> WT[Nrows][Kpad] bf16 ----------------
__global__ __launch_bounds__(256) void wtconv_kernel(
    const float* __restrict__ W, __bf16* __restrict__ WT,
    int K, int Nin, int Kpad, int Nrows)
{
  int i = blockIdx.x * 256 + threadIdx.x;
  if (i >= Nrows * Kpad) return;
  int k = i % Kpad;
  int n = i / Kpad;
  float v = (k < K && n < Nin) ? W[(size_t)k * Nin + n] : 0.f;
  WT[i] = (__bf16)v;
}

// ---------------- generic WMMA GEMM with async-to-LDS B staging ----------------
// C = act(A @ WT^T + bias [+ resid]).  A: bf16 [M x lda] (streamed from global),
// WT: bf16 [Ncols x ldb] (row n = column n of W; Ncols multiple of NB*16, zero-padded).
// grid = (M/128, Ncols/(NB*16)), block = 256 (8 waves); each wave: 16 rows x NB*16 cols.
// B tile (NB*16 rows x 32 k = NB*1KB) is double-buffer staged into LDS with
// GLOBAL_LOAD_ASYNC_TO_LDS_B128 (ASYNCcnt) and shared by all 8 waves.
template <int NB>
__global__ __launch_bounds__(256) void gemm_bf16_kernel(
    const __bf16* __restrict__ A, int lda,
    const __bf16* __restrict__ Bt, int ldb,
    const float* __restrict__ bias,
    const float* __restrict__ resid, int ldr,
    float* __restrict__ C, int ldc,
    int Ksteps, int Nvalid, int relu)
{
  __shared__ __bf16 bsm[2][NB * 16][32];   // NB=4: 8 KB
  int tid = threadIdx.x;
  int lane = tid & 31;
  int wv = tid >> 5;
  int mbase = blockIdx.x * 128 + wv * 16;
  int nbase0 = blockIdx.y * (NB * 16);
  int l16 = lane & 15;
  int ksa = (lane >> 4) * 16;
  int ksb = (lane >> 4) * 16;
  const __bf16* ap = A + (size_t)(mbase + l16) * lda;

  // staging map: one 16B chunk per thread; NB*16 rows x 4 chunks = NB*64 units
  bool do_stage = tid < NB * 64;
  int srow = tid >> 2;
  int schunk = tid & 3;
  const __bf16* sgp = Bt + (size_t)(nbase0 + srow) * ldb + schunk * 8;
  // low 32 bits of the generic pointer to __shared__ == LDS byte offset (ISA 10.2)
  uint32_t lds0 = (uint32_t)(uintptr_t)&bsm[0][srow][schunk * 8];
  uint32_t lds1 = (uint32_t)(uintptr_t)&bsm[1][srow][schunk * 8];

  v8f acc[NB] = {};

  // prologue: stage k-step 0 into buffer 0
  if (do_stage) {
    uint64_t ga = (uint64_t)(uintptr_t)(sgp);
    asm volatile("global_load_async_to_lds_b128 %0, %1, off"
                 :: "v"(lds0), "v"(ga) : "memory");
  }

  for (int ks = 0; ks < Ksteps; ++ks) {
    bool more = (ks + 1 < Ksteps);
    if (more && do_stage) {
      uint64_t ga = (uint64_t)(uintptr_t)(sgp + (size_t)(ks + 1) * 32);
      uint32_t lo = ((ks + 1) & 1) ? lds1 : lds0;
      asm volatile("global_load_async_to_lds_b128 %0, %1, off"
                   :: "v"(lo), "v"(ga) : "memory");
    }
    if (more) asm volatile("s_wait_asynccnt 1" ::: "memory");
    else      asm volatile("s_wait_asynccnt 0" ::: "memory");
    __syncthreads();

    int ka = ks * 32 + ksa;
    v8bf a0 = *(const v8bf*)(ap + ka);
    v8bf a1 = *(const v8bf*)(ap + ka + 8);
    v16bf a = cat8(a0, a1);
    if (more) __builtin_prefetch(ap + ka + 128, 0, 1);   // global_prefetch_b8
    int buf = ks & 1;
    #pragma unroll
    for (int j = 0; j < NB; ++j) {
      v16bf b = *(const v16bf*)(&bsm[buf][j * 16 + l16][ksb]);
      acc[j] = __builtin_amdgcn_wmma_f32_16x16x32_bf16(false, a, false, b, (short)0, acc[j], false, false);
    }
    __syncthreads();   // protect buffer about to be overwritten
  }

  int rbase = mbase + (lane >> 4) * 8;
  #pragma unroll
  for (int j = 0; j < NB; ++j) {
    int col = nbase0 + j * 16 + l16;
    if (col < Nvalid) {
      float bv = bias ? bias[col] : 0.f;
      #pragma unroll
      for (int r = 0; r < 8; ++r) {
        int rr = rbase + r;
        float v = acc[j][r] + bv;
        if (resid) v += resid[(size_t)rr * ldr + col];
        if (relu) v = fmaxf(v, 0.f);
        C[(size_t)rr * ldc + col] = v;
      }
    }
  }
}

// ---------------- pack Q/K rows (padded) ----------------
__global__ __launch_bounds__(256) void pack_qk_kernel(
    const float* __restrict__ src, __bf16* __restrict__ dst)
{
  int i = blockIdx.x * 256 + threadIdx.x;
  if (i >= BT * H * NP * DHP) return;
  int kk = i % DHP;
  int r = (i / DHP) % NP;
  int hh = (i / (DHP * NP)) % H;
  int bt = i / (DHP * NP * H);
  float v = 0.f;
  if (r < NN && kk < DH) v = src[((size_t)bt * NN + r) * DM + hh * DH + kk];
  dst[i] = (__bf16)v;
}

// ---------------- pack V transposed: Vt[bt][h][d][node] ----------------
__global__ __launch_bounds__(256) void pack_vt_kernel(
    const float* __restrict__ src, __bf16* __restrict__ dst)
{
  int i = blockIdx.x * 256 + threadIdx.x;
  if (i >= BT * H * DHT * NP) return;
  int cnode = i % NP;
  int dr = (i / NP) % DHT;
  int hh = (i / (NP * DHT)) % H;
  int bt = i / (NP * DHT * H);
  float v = 0.f;
  if (cnode < NN && dr < DH) v = src[((size_t)bt * NN + cnode) * DM + hh * DH + dr];
  dst[i] = (__bf16)v;
}

// ---------------- fused attention: scores(WMMA) -> softmax(LDS) -> ctx(WMMA) ----------------
// grid = (25 row-tiles, H, BT), block = 256
__global__ __launch_bounds__(256) void attn_kernel(
    const __bf16* __restrict__ Qp, const __bf16* __restrict__ Kp,
    const __bf16* __restrict__ Vt, const int* __restrict__ spd,
    const float* __restrict__ gamma, float* __restrict__ ctx)
{
  __shared__ float sc[16][NP];
  int mt = blockIdx.x, hh = blockIdx.y, bt = blockIdx.z;
  int lane = threadIdx.x & 31, wv = threadIdx.x >> 5;
  int mbase = mt * 16;
  const __bf16* Qb = Qp + (size_t)(bt * H + hh) * NP * DHP;
  const __bf16* Kb = Kp + (size_t)(bt * H + hh) * NP * DHP;
  const __bf16* Vb = Vt + (size_t)(bt * H + hh) * DHT * NP;
  float g = gamma[hh];
  int l16 = lane & 15;
  int ksa = (lane >> 4) * 16;
  int ksb = (lane >> 4) * 16;
  const __bf16* qrow = Qb + (size_t)(mbase + l16) * DHP;

  // ---- phase 1: score tiles ----
  for (int nt = wv; nt < 25; nt += 8) {
    int nbase = nt * 16;
    const __bf16* krow = Kb + (size_t)(nbase + l16) * DHP;
    v8f c = {};
    #pragma unroll
    for (int ks = 0; ks < 2; ++ks) {
      int ka = ks * 32 + ksa;
      v8bf a0 = *(const v8bf*)(qrow + ka);
      v8bf a1 = *(const v8bf*)(qrow + ka + 8);
      v16bf a = cat8(a0, a1);
      v16bf b = *(const v16bf*)(krow + ks * 32 + ksb);
      c = __builtin_amdgcn_wmma_f32_16x16x32_bf16(false, a, false, b, (short)0, c, false, false);
    }
    int rb = (lane >> 4) * 8;
    #pragma unroll
    for (int r = 0; r < 8; ++r) {
      int lr = rb + r;
      int gr = mbase + lr;
      int gc = nbase + l16;
      float val = c[r] * SCALE;
      if (hh > 0 && spd[gr * NN + gc] == hh) val += g;
      sc[lr][gc] = val;
    }
  }
  __syncthreads();

  // ---- phase 2: softmax (2 rows per wave) ----
  for (int r = wv * 2; r < wv * 2 + 2; ++r) {
    float m = -1e30f;
    for (int c2 = lane; c2 < NN; c2 += 32) m = fmaxf(m, sc[r][c2]);
    for (int o = 16; o; o >>= 1) m = fmaxf(m, __shfl_xor(m, o));
    float ssum = 0.f;
    for (int c2 = lane; c2 < NN; c2 += 32) {
      float e = __expf(sc[r][c2] - m);
      sc[r][c2] = e;
      ssum += e;
    }
    for (int o = 16; o; o >>= 1) ssum += __shfl_xor(ssum, o);
    float inv = 1.f / ssum;
    for (int c2 = lane; c2 < NN; c2 += 32) sc[r][c2] *= inv;
    if (lane < 16) sc[r][NN + lane] = 0.f;  // zero the K-pad columns
  }
  __syncthreads();

  // ---- phase 3: context (3 d-tiles on waves 0..2) ----
  if (wv < 3) {
    int nbase = wv * 16;
    const __bf16* vrow = Vb + (size_t)(nbase + l16) * NP;
    v8f c = {};
    for (int ks = 0; ks < NP / 32; ++ks) {
      int ka = ks * 32 + ksa;
      v16bf a;
      #pragma unroll
      for (int q = 0; q < 8; ++q) {
        a[q] = (__bf16)sc[l16][ka + q];
        a[q + 8] = (__bf16)sc[l16][ka + 8 + q];
      }
      v16bf b = *(const v16bf*)(vrow + ks * 32 + ksb);
      c = __builtin_amdgcn_wmma_f32_16x16x32_bf16(false, a, false, b, (short)0, c, false, false);
    }
    int dcol = nbase + l16;
    if (dcol < DH) {
      int rb = (lane >> 4) * 8;
      #pragma unroll
      for (int r = 0; r < 8; ++r) {
        int gr = mbase + rb + r;
        ctx[((size_t)bt * NN + gr) * DM + hh * DH + dcol] = c[r];
      }
    }
  }
}

// ---------------- layernorm over last dim (DM) ----------------
__global__ __launch_bounds__(256) void ln_kernel(
    const float* __restrict__ X, const float* __restrict__ g,
    const float* __restrict__ bta, float* __restrict__ Y, int Mtot)
{
  int lane = threadIdx.x & 31, wv = threadIdx.x >> 5;
  int row = blockIdx.x * 8 + wv;
  if (row >= Mtot) return;
  const float* xr = X + (size_t)row * DM;
  float s = 0.f;
  for (int c = lane; c < DM; c += 32) s += xr[c];
  for (int o = 16; o; o >>= 1) s += __shfl_xor(s, o);
  float mean = s * (1.f / DM);
  float v = 0.f;
  for (int c = lane; c < DM; c += 32) { float d = xr[c] - mean; v += d * d; }
  for (int o = 16; o; o >>= 1) v += __shfl_xor(v, o);
  float rstd = rsqrtf(v * (1.f / DM) + EPS);
  float* yr = Y + (size_t)row * DM;
  for (int c = lane; c < DM; c += 32) yr[c] = (xr[c] - mean) * rstd * g[c] + bta[c];
}

// ---------------- gather for output projection: h -> A[b*N+n][t*DM+d] bf16 ----------------
__global__ __launch_bounds__(256) void gather_out_kernel(
    const float* __restrict__ h, __bf16* __restrict__ outA)
{
  int i = blockIdx.x * 256 + threadIdx.x;
  if (i >= B * NN * KOUT) return;
  int k = i % KOUT;
  int rr = i / KOUT;
  int n = rr % NN;
  int b = rr / NN;
  int t = k / DM;
  int d = k % DM;
  outA[i] = (__bf16)h[(((size_t)b * T + t) * NN + n) * DM + d];
}

// ---------------- output remap (B,N,9) -> (B,3,N,3) ----------------
__global__ __launch_bounds__(256) void out_remap_kernel(
    const float* __restrict__ tmp, float* __restrict__ out)
{
  int i = blockIdx.x * 256 + threadIdx.x;
  if (i >= B * NN * OUT_STEPS * OUT_DIM) return;
  int j = i % 9;
  int n = (i / 9) % NN;
  int b = i / (9 * NN);
  int s = j / 3, o = j % 3;
  out[(((size_t)b * OUT_STEPS + s) * NN + n) * OUT_DIM + o] = tmp[i];
}

// ---------------- host orchestration ----------------
static inline int cdiv(int a, int b) { return (a + b - 1) / b; }

extern "C" void kernel_launch(void* const* d_in, const int* in_sizes, int n_in,
                              void* d_out, int out_size, void* d_ws, size_t ws_size,
                              hipStream_t stream) {
  const float* x       = (const float*)d_in[0];
  const int*   tod     = (const int*)  d_in[1];
  const int*   dow     = (const int*)  d_in[2];
  const int*   spd     = (const int*)  d_in[3];
  const float* in_W    = (const float*)d_in[4];
  const float* in_b    = (const float*)d_in[5];
  const float* tod_emb = (const float*)d_in[6];
  const float* dow_emb = (const float*)d_in[7];
  const float* adp     = (const float*)d_in[8];
  const float* dlin_Ws = (const float*)d_in[9];
  const float* dlin_bs = (const float*)d_in[10];
  const float* dlin_Wt = (const float*)d_in[11];
  const float* dlin_bt = (const float*)d_in[12];
  const float* Wq = (const float*)d_in[13]; const float* bq = (const float*)d_in[14];
  const float* Wk = (const float*)d_in[15]; const float* bk = (const float*)d_in[16];
  const float* Wv = (const float*)d_in[17]; const float* bv = (const float*)d_in[18];
  const float* Wo = (const float*)d_in[19]; const float* bo = (const float*)d_in[20];
  const float* gamma = (const float*)d_in[21];
  const float* ffW1 = (const float*)d_in[22]; const float* ffb1 = (const float*)d_in[23];
  const float* ffW2 = (const float*)d_in[24]; const float* ffb2 = (const float*)d_in[25];
  const float* ln1g = (const float*)d_in[26]; const float* ln1b = (const float*)d_in[27];
  const float* ln2g = (const float*)d_in[28]; const float* ln2b = (const float*)d_in[29];
  const float* outW = (const float*)d_in[30]; const float* outb = (const float*)d_in[31];

  // ---- workspace carve (~240 MB; all offsets 256B aligned) ----
  char* base = (char*)d_ws;
  size_t off = 0;
  auto carve = [&](size_t bytes) -> char* {
    off = (off + 255) & ~(size_t)255;
    char* r = base + off;
    off += bytes;
    return r;
  };
  float* h    = (float*)carve((size_t)M_ROWS * DM * 4);
  float* qb   = (float*)carve((size_t)M_ROWS * DM * 4);  // also ctx target
  float* kb   = (float*)carve((size_t)M_ROWS * DM * 4);
  float* vb   = (float*)carve((size_t)M_ROWS * DM * 4);
  float* ffb  = (float*)carve((size_t)M_ROWS * FF * 4);  // Wo-out / ff1-out / ff2-out
  __bf16* abf = (__bf16*)carve((size_t)M_ROWS * FF * 2); // activation bf16 (also outA)
  __bf16* Qp  = (__bf16*)carve((size_t)BT * H * NP * DHP * 2);
  __bf16* Kp  = (__bf16*)carve((size_t)BT * H * NP * DHP * 2);
  __bf16* Vt  = (__bf16*)carve((size_t)BT * H * DHT * NP * 2);
  __bf16* wtq = (__bf16*)carve((size_t)DMP * DMP * 2);   // 192 rows (zero padded)
  __bf16* wtk = (__bf16*)carve((size_t)DMP * DMP * 2);
  __bf16* wtv = (__bf16*)carve((size_t)DMP * DMP * 2);
  __bf16* wto = (__bf16*)carve((size_t)DMP * DMP * 2);
  __bf16* wtf1 = (__bf16*)carve((size_t)FF * DMP * 2);   // 256 rows
  __bf16* wtf2 = (__bf16*)carve((size_t)DMP * FF * 2);   // 192 rows x 256
  __bf16* outWT = (__bf16*)carve((size_t)16 * KOUT * 2);
  float* outtmp = (float*)carve((size_t)B * NN * 9 * 4);
  __bf16* outA = abf; // reuse: abf dead after last FF gemm
  float* ctx = qb;    // reuse: raw q dead after pack

  // ---- embedding ----
  embed_kernel<<<cdiv(M_ROWS * DM, 256), 256, 0, stream>>>(
      x, tod, dow, in_W, in_b, tod_emb, dow_emb, adp, h);

  dim3 gDM(M_ROWS / 128, DMP / 64);   // (300, 3)  -> 192 padded cols, Nvalid=176
  dim3 gFF(M_ROWS / 128, FF / 64);    // (300, 4)

  for (int l = 0; l < L; ++l) {
    // DLinear (in place)
    dlinear_kernel<<<cdiv(B * NN * DM, 256), 256, 0, stream>>>(
        h, dlin_Ws + l * T * T, dlin_bs + l * T, dlin_Wt + l * T * T, dlin_bt + l * T);

    // weights -> bf16 transposed (rows zero-padded to the NB*16 grid)
    wtconv_kernel<<<cdiv(DMP * DMP, 256), 256, 0, stream>>>(Wq + (size_t)l * DM * DM, wtq, DM, DM, DMP, DMP);
    wtconv_kernel<<<cdiv(DMP * DMP, 256), 256, 0, stream>>>(Wk + (size_t)l * DM * DM, wtk, DM, DM, DMP, DMP);
    wtconv_kernel<<<cdiv(DMP * DMP, 256), 256, 0, stream>>>(Wv + (size_t)l * DM * DM, wtv, DM, DM, DMP, DMP);
    wtconv_kernel<<<cdiv(DMP * DMP, 256), 256, 0, stream>>>(Wo + (size_t)l * DM * DM, wto, DM, DM, DMP, DMP);
    wtconv_kernel<<<cdiv(FF * DMP, 256), 256, 0, stream>>>(ffW1 + (size_t)l * DM * FF, wtf1, DM, FF, DMP, FF);
    wtconv_kernel<<<cdiv(DMP * FF, 256), 256, 0, stream>>>(ffW2 + (size_t)l * FF * DM, wtf2, FF, DM, FF, DMP);

    // h -> bf16 (K padded to 192)
    convpad_kernel<<<cdiv(M_ROWS * DMP, 256), 256, 0, stream>>>(h, abf, DM, DMP, M_ROWS * DMP);

    // QKV projections (WMMA, async B staging)
    gemm_bf16_kernel<4><<<gDM, 256, 0, stream>>>(abf, DMP, wtq, DMP, bq + l * DM, nullptr, 0, qb, DM, DMP / 32, DM, 0);
    gemm_bf16_kernel<4><<<gDM, 256, 0, stream>>>(abf, DMP, wtk, DMP, bk + l * DM, nullptr, 0, kb, DM, DMP / 32, DM, 0);
    gemm_bf16_kernel<4><<<gDM, 256, 0, stream>>>(abf, DMP, wtv, DMP, bv + l * DM, nullptr, 0, vb, DM, DMP / 32, DM, 0);

    // pack padded bf16 Q/K and transposed V
    pack_qk_kernel<<<cdiv(BT * H * NP * DHP, 256), 256, 0, stream>>>(qb, Qp);
    pack_qk_kernel<<<cdiv(BT * H * NP * DHP, 256), 256, 0, stream>>>(kb, Kp);
    pack_vt_kernel<<<cdiv(BT * H * DHT * NP, 256), 256, 0, stream>>>(vb, Vt);

    // fused attention (WMMA scores + softmax + WMMA ctx)
    attn_kernel<<<dim3(NN / 16, H, BT), 256, 0, stream>>>(Qp, Kp, Vt, spd, gamma + l * H, ctx);

    // ctx @ Wo + bo + resid(h) -> ffb ; LN1 -> h
    convpad_kernel<<<cdiv(M_ROWS * DMP, 256), 256, 0, stream>>>(ctx, abf, DM, DMP, M_ROWS * DMP);
    gemm_bf16_kernel<4><<<gDM, 256, 0, stream>>>(abf, DMP, wto, DMP, bo + l * DM, h, DM, ffb, DM, DMP / 32, DM, 0);
    ln_kernel<<<M_ROWS / 8, 256, 0, stream>>>(ffb, ln1g + l * DM, ln1b + l * DM, h, M_ROWS);

    // FFN: relu(h@W1+b1)@W2+b2 + resid(h) ; LN2 -> h
    convpad_kernel<<<cdiv(M_ROWS * DMP, 256), 256, 0, stream>>>(h, abf, DM, DMP, M_ROWS * DMP);
    gemm_bf16_kernel<4><<<gFF, 256, 0, stream>>>(abf, DMP, wtf1, DMP, ffb1 + l * FF, nullptr, 0, ffb, FF, DMP / 32, FF, 1);
    convpad_kernel<<<cdiv(M_ROWS * FF, 256), 256, 0, stream>>>(ffb, abf, FF, FF, M_ROWS * FF);
    gemm_bf16_kernel<4><<<gDM, 256, 0, stream>>>(abf, FF, wtf2, FF, ffb2 + l * DM, h, DM, ffb, DM, FF / 32, DM, 0);
    ln_kernel<<<M_ROWS / 8, 256, 0, stream>>>(ffb, ln2g + l * DM, ln2b + l * DM, h, M_ROWS);
  }

  // ---- output projection ----
  wtconv_kernel<<<cdiv(16 * KOUT, 256), 256, 0, stream>>>(outW, outWT, KOUT, 9, KOUT, 16);
  gather_out_kernel<<<cdiv(B * NN * KOUT, 256), 256, 0, stream>>>(h, outA);
  gemm_bf16_kernel<1><<<dim3(B * NN / 128, 1), 256, 0, stream>>>(
      outA, KOUT, outWT, KOUT, outb, nullptr, 0, outtmp, 9, KOUT / 32, 9, 0);
  out_remap_kernel<<<cdiv(B * NN * 9, 256), 256, 0, stream>>>(outtmp, (float*)d_out);
}
